// MultiHeadAttn_66443144069849
// MI455X (gfx1250) — matrix operand order repaired and checked by reading, not compiled, not run on hardware
//
#include <hip/hip_runtime.h>

// ---------------------------------------------------------------------------
// Types for CDNA5 WMMA (wave32, 16x16x32 bf16 -> f32)
// ---------------------------------------------------------------------------
typedef __bf16 bf16_t;
typedef __attribute__((ext_vector_type(16))) __bf16 v16bf;
typedef __attribute__((ext_vector_type(8)))  __bf16 v8bf;
typedef __attribute__((ext_vector_type(8)))  float  v8f;

#define WMMA_BF16(a, b, c) \
  __builtin_amdgcn_wmma_f32_16x16x32_bf16(false, (a), false, (b), (short)0, (c), false, false)

// Problem constants
constexpr int Bsz  = 4;
constexpr int Nseq = 2048;
constexpr int Din  = 512;
constexpr int Emb  = 512;
constexpr int H    = 8;
constexpr int CH   = 64;
constexpr int BN   = Bsz * Nseq;          // 8192

// Workspace layout (bytes)
constexpr size_t XBF_OFF  = 0;                                    // 3 x [BN][Din] bf16
constexpr size_t XBF_ONE  = (size_t)BN * Din * 2;                 // 8 MiB each
constexpr size_t WT_OFF   = XBF_OFF + 3 * XBF_ONE;                // 3 x [Emb][Din] bf16 (W^T)
constexpr size_t WT_ONE   = (size_t)Emb * Din * 2;
constexpr size_t Q_OFF    = WT_OFF + 3 * WT_ONE;                  // [32][2048][64] bf16
constexpr size_t HEAD_ONE = (size_t)Bsz * H * Nseq * CH * 2;      // 8 MiB
constexpr size_t K_OFF    = Q_OFF + HEAD_ONE;                     // [32][2048][64] bf16
constexpr size_t V_OFF    = K_OFF + HEAD_ONE;                     // [32][64][2048] bf16 (V^T)

// ---------------------------------------------------------------------------
// Fragment loaders (this code's contiguous k order, same for A and B)
// A (16x32, MxK): lane m<16 -> row m; per-lane K = kh*16 + [0..15] (contiguous)
// B (32x16, KxN): lane n<16 -> col n; per-lane K = kh*16 + [0..15] (contiguous)
// ---------------------------------------------------------------------------
union FragBF {
  v16bf v;
  v8bf  h[2];
  unsigned short s[16];
};

__device__ inline v16bf load_frag_a(const bf16_t* base, int ld, int c0) {
  int lane = threadIdx.x & 31;
  int m = lane & 15, kh = lane >> 4;
  const bf16_t* p = base + (size_t)m * ld + c0 + kh * 16;
  FragBF f;
  f.h[0] = *(const v8bf*)(p);
  f.h[1] = *(const v8bf*)(p + 8);
  return f.v;
}

__device__ inline v16bf load_frag_b(const bf16_t* base, int ld, int c0) {
  int lane = threadIdx.x & 31;
  int n = lane & 15, kh = lane >> 4;
  const bf16_t* p = base + (size_t)n * ld + c0 + kh * 16;
  FragBF f;
  f.h[0] = *(const v8bf*)(p);
  f.h[1] = *(const v8bf*)(p + 8);
  return f.v;
}

__device__ inline unsigned short f2bf(float x) {
  bf16_t h = (bf16_t)x;
  return __builtin_bit_cast(unsigned short, h);
}

// ---------------------------------------------------------------------------
// Kernel 0: convert q,k,v fp32 -> bf16 staging (packed 16B stores)
// ---------------------------------------------------------------------------
__global__ void __launch_bounds__(256) cvt_inputs_kernel(
    const float* __restrict__ q, const float* __restrict__ k,
    const float* __restrict__ v, bf16_t* __restrict__ xbf) {
  int tid = blockIdx.x * blockDim.x + threadIdx.x;   // 0 .. 1572863
  int which = tid / 524288;                           // 524288 threads per tensor
  int idx = (tid - which * 524288) * 8;
  const float* src = (which == 0) ? q : ((which == 1) ? k : v);
  const float4* s = (const float4*)(src + idx);
  float4 a = s[0], b = s[1];
  v8bf o;
  o[0] = (bf16_t)a.x; o[1] = (bf16_t)a.y; o[2] = (bf16_t)a.z; o[3] = (bf16_t)a.w;
  o[4] = (bf16_t)b.x; o[5] = (bf16_t)b.y; o[6] = (bf16_t)b.z; o[7] = (bf16_t)b.w;
  *(v8bf*)(xbf + (size_t)which * BN * Din + idx) = o;
}

// ---------------------------------------------------------------------------
// Kernel 1: convert + transpose weights: Wt[e][d] = (bf16) W[d][e]
// ---------------------------------------------------------------------------
__global__ void __launch_bounds__(256) cvt_weights_kernel(
    const float* __restrict__ wq, const float* __restrict__ wk,
    const float* __restrict__ wv, bf16_t* __restrict__ wt) {
  int tid = blockIdx.x * blockDim.x + threadIdx.x;   // 0 .. 786431
  int which = tid >> 18;                              // / 262144
  int r = tid & 262143;
  int d = r >> 9;
  int e = r & 511;
  const float* w = (which == 0) ? wq : ((which == 1) ? wk : wv);
  wt[(size_t)which * Emb * Din + (size_t)e * Din + d] = (bf16_t)w[(size_t)d * Emb + e];
}

// ---------------------------------------------------------------------------
// Kernel 2: projections. One wave -> one 32(M) x 64(E) output tile, K=512.
// Ping-pong pipelined (two static fragment buffer sets, no register rotation):
//   load B(2i+1) | 8 WMMA on A(2i) | load A(2i+2) | 8 WMMA on B(2i+1)
// proj 0/1 (Q,K): store row-major [bh][n][64].  proj 2 (V): store V^T [bh][64][n].
// ---------------------------------------------------------------------------
__global__ void __launch_bounds__(128) proj_kernel(
    const bf16_t* __restrict__ xbf, const bf16_t* __restrict__ wt,
    const float* __restrict__ bq, const float* __restrict__ bk,
    const float* __restrict__ bv,
    bf16_t* __restrict__ Qh, bf16_t* __restrict__ Kh, bf16_t* __restrict__ Vt) {
  int warp = threadIdx.x >> 5;
  int w = blockIdx.x * 4 + warp;        // 0..6143
  int proj = w >> 11;                   // 2048 waves per projection
  int rem = w & 2047;
  int mtile = rem >> 3;                 // 256 M tiles of 32 rows
  int eg = rem & 7;                     // 8 groups of 64 output cols
  int m0 = mtile * 32, e0 = eg * 64;

  const bf16_t* X0 = xbf + (size_t)proj * BN * Din + (size_t)m0 * Din;
  const bf16_t* X1 = X0 + (size_t)16 * Din;
  const bf16_t* W = wt + (size_t)proj * Emb * Din;
  const bf16_t* Wr0 = W + (size_t)(e0)      * Din;
  const bf16_t* Wr1 = W + (size_t)(e0 + 16) * Din;
  const bf16_t* Wr2 = W + (size_t)(e0 + 32) * Din;
  const bf16_t* Wr3 = W + (size_t)(e0 + 48) * Din;
  const float* bias = (proj == 0) ? bq : ((proj == 1) ? bk : bv);

  v8f acc[8] = {};                      // [t] rows 0..15, [4+t] rows 16..31

  // prologue: buffer A <- chunk 0
  v16bf a0A = load_frag_a(X0, Din, 0);
  v16bf a1A = load_frag_a(X1, Din, 0);
  v16bf bA[4];
  bA[0] = load_frag_b(Wr0, Din, 0);
  bA[1] = load_frag_b(Wr1, Din, 0);
  bA[2] = load_frag_b(Wr2, Din, 0);
  bA[3] = load_frag_b(Wr3, Din, 0);

  // 16 chunks of K=32; pairs (2i, 2i+1) for i=0..6, chunks 14,15 in tail
  for (int i = 0; i < 7; i++) {
    int kB = i * 64 + 32;
    int kA = i * 64 + 64;
    // buffer B <- chunk 2i+1 (latency covered by A's WMMAs)
    v16bf a0B = load_frag_a(X0, Din, kB);
    v16bf a1B = load_frag_a(X1, Din, kB);
    v16bf bB[4];
    bB[0] = load_frag_b(Wr0, Din, kB);
    bB[1] = load_frag_b(Wr1, Din, kB);
    bB[2] = load_frag_b(Wr2, Din, kB);
    bB[3] = load_frag_b(Wr3, Din, kB);
#pragma unroll
    for (int t = 0; t < 4; t++) {
      acc[t]     = WMMA_BF16(a0A, bA[t], acc[t]);
      acc[4 + t] = WMMA_BF16(a1A, bA[t], acc[4 + t]);
    }
    // buffer A <- chunk 2i+2 (latency covered by B's WMMAs + backedge)
    a0A = load_frag_a(X0, Din, kA);
    a1A = load_frag_a(X1, Din, kA);
    bA[0] = load_frag_b(Wr0, Din, kA);
    bA[1] = load_frag_b(Wr1, Din, kA);
    bA[2] = load_frag_b(Wr2, Din, kA);
    bA[3] = load_frag_b(Wr3, Din, kA);
#pragma unroll
    for (int t = 0; t < 4; t++) {
      acc[t]     = WMMA_BF16(a0B, bB[t], acc[t]);
      acc[4 + t] = WMMA_BF16(a1B, bB[t], acc[4 + t]);
    }
  }
  {
    // tail: A holds chunk 14 (k=448); load chunk 15 (k=480) into B
    v16bf a0B = load_frag_a(X0, Din, 480);
    v16bf a1B = load_frag_a(X1, Din, 480);
    v16bf bB[4];
    bB[0] = load_frag_b(Wr0, Din, 480);
    bB[1] = load_frag_b(Wr1, Din, 480);
    bB[2] = load_frag_b(Wr2, Din, 480);
    bB[3] = load_frag_b(Wr3, Din, 480);
#pragma unroll
    for (int t = 0; t < 4; t++) {
      acc[t]     = WMMA_BF16(a0A, bA[t], acc[t]);
      acc[4 + t] = WMMA_BF16(a1A, bA[t], acc[4 + t]);
    }
#pragma unroll
    for (int t = 0; t < 4; t++) {
      acc[t]     = WMMA_BF16(a0B, bB[t], acc[t]);
      acc[4 + t] = WMMA_BF16(a1B, bB[t], acc[4 + t]);
    }
  }

  int lane = threadIdx.x & 31;
  int col = lane & 15, mb = (lane >> 4) * 8;
  int b_ = m0 >> 11;                    // batch index (32-row tile never crosses)
  int nb = (m0 & 2047) + mb;

#pragma unroll
  for (int t = 0; t < 4; t++) {
    int e = e0 + t * 16 + col;
    float bb = bias[e];
    int h = e >> 6, c = e & 63;
    if (proj < 2) {
      bf16_t* dst = ((proj == 0) ? Qh : Kh) + (size_t)(b_ * H + h) * Nseq * CH + c;
#pragma unroll
      for (int r = 0; r < 8; r++) {
        dst[(size_t)(nb + r) * CH]      = (bf16_t)(acc[t][r] + bb);
        dst[(size_t)(nb + 16 + r) * CH] = (bf16_t)(acc[4 + t][r] + bb);
      }
    } else {
      v8bf p0, p1;
#pragma unroll
      for (int r = 0; r < 8; r++) {
        p0[r] = (bf16_t)(acc[t][r] + bb);
        p1[r] = (bf16_t)(acc[4 + t][r] + bb);
      }
      bf16_t* dst = Vt + ((size_t)(b_ * H + h) * CH + c) * Nseq;
      *(v8bf*)(dst + nb)      = p0;
      *(v8bf*)(dst + nb + 16) = p1;
    }
  }
}

// ---------------------------------------------------------------------------
// Kernel 3: flash attention, transposed: S^T = K Q^T, O^T = V^T P^T.
// One wave owns 16 queries; online softmax stats are per-lane scalars.
// All fragments are produced/consumed within one iteration (no rotation).
// ---------------------------------------------------------------------------
__global__ void __launch_bounds__(128) attn_kernel(
    const bf16_t* __restrict__ Qh, const bf16_t* __restrict__ Kh,
    const bf16_t* __restrict__ Vt, float* __restrict__ out) {
  int warp = threadIdx.x >> 5;
  int w = blockIdx.x * 4 + warp;        // 0..4095
  int bh = w >> 7;                      // 32 batch-heads
  int q0 = (w & 127) * 16;              // 128 query tiles
  int lane = threadIdx.x & 31;
  int col = lane & 15, kh = lane >> 4;

  const bf16_t* Qr = Qh + ((size_t)bh * Nseq + q0) * CH;
  const bf16_t* Kb = Kh + (size_t)bh * Nseq * CH;
  const bf16_t* Vb = Vt + (size_t)bh * CH * Nseq;

  // Q as WMMA B operand (lane = query column), resident for whole k loop
  v16bf qb0 = load_frag_b(Qr, CH, 0);
  v16bf qb1 = load_frag_b(Qr, CH, 32);

  float mrun = -3.0e38f, lrun = 0.0f;
  v8f O[4] = {};
  const float sc = 0.125f;              // 1/sqrt(64)

  for (int k0 = 0; k0 < Nseq; k0 += 32) {
    // all chunk loads issued up-front
    v16bf ka00 = load_frag_a(Kb + (size_t)k0 * CH, CH, 0);
    v16bf ka01 = load_frag_a(Kb + (size_t)k0 * CH, CH, 32);
    v16bf ka10 = load_frag_a(Kb + (size_t)(k0 + 16) * CH, CH, 0);
    v16bf ka11 = load_frag_a(Kb + (size_t)(k0 + 16) * CH, CH, 32);
    v16bf va[4];
#pragma unroll
    for (int t = 0; t < 4; t++)
      va[t] = load_frag_a(Vb + (size_t)(t * 16) * Nseq, Nseq, k0);

    // prefetch next chunk: K chunk = 4KB = 32 lines (one per lane);
    // V chunk = 64 rows, 64B each (2 prefetches, rows lane and lane+32)
    const char* kpf = (const char*)(Kb + (size_t)(k0 + 32) * CH) + lane * 128;
    __builtin_prefetch(kpf, 0, 3);
    const char* vpf = (const char*)Vb + (size_t)lane * (Nseq * 2) + (size_t)(k0 + 32) * 2;
    __builtin_prefetch(vpf, 0, 3);
    __builtin_prefetch(vpf + (size_t)32 * Nseq * 2, 0, 3);

    // S^T tiles for keys [k0, k0+16) and [k0+16, k0+32)
    v8f z = {};
    v8f s0 = WMMA_BF16(ka00, qb0, z);
    s0 = WMMA_BF16(ka01, qb1, s0);
    v8f s1 = WMMA_BF16(ka10, qb0, z);
    s1 = WMMA_BF16(ka11, qb1, s1);

    // online softmax over the k (row) direction; q is fixed per lane
    float mloc = -3.0e38f;
#pragma unroll
    for (int r = 0; r < 8; r++) {
      s0[r] *= sc; s1[r] *= sc;
      mloc = fmaxf(mloc, fmaxf(s0[r], s1[r]));
    }
    mloc = fmaxf(mloc, __shfl_xor(mloc, 16, 32));
    float mnew = fmaxf(mrun, mloc);
    float alpha = __expf(mrun - mnew);

    // P^T -> WMMA B operand: lane needs k = kh*16 + [0..15]; D gave kh*8-split
    FragBF pf;
    float rs = 0.0f;
#pragma unroll
    for (int r = 0; r < 8; r++) {
      float p0 = __expf(s0[r] - mnew);  // k = kh*8 + r
      float p1 = __expf(s1[r] - mnew);  // k = 16 + kh*8 + r
      rs += p0 + p1;
      unsigned int pk = ((unsigned int)f2bf(p1) << 16) | (unsigned int)f2bf(p0);
      unsigned int px = __shfl_xor(pk, 16, 32);
      pf.s[r]     = kh ? (unsigned short)(px >> 16) : (unsigned short)(pk & 0xffffu);
      pf.s[8 + r] = kh ? (unsigned short)(pk >> 16) : (unsigned short)(px & 0xffffu);
    }
    rs += __shfl_xor(rs, 16, 32);
    lrun = lrun * alpha + rs;
    mrun = mnew;

#pragma unroll
    for (int t = 0; t < 4; t++)
#pragma unroll
      for (int r = 0; r < 8; r++) O[t][r] *= alpha;

    // O^T += V^T(c-tile, k-chunk) x P^T
#pragma unroll
    for (int t = 0; t < 4; t++)
      O[t] = WMMA_BF16(va[t], pf.v, O[t]);
  }

  float inv = 1.0f / lrun;
  int b_ = bh >> 3, h = bh & 7;
  float* orow = out + ((size_t)b_ * Nseq + q0 + col) * Emb + h * CH;
#pragma unroll
  for (int t = 0; t < 4; t++)
#pragma unroll
    for (int r = 0; r < 8; r++)
      orow[t * 16 + kh * 8 + r] = O[t][r] * inv;
}

// ---------------------------------------------------------------------------
// Launch
// ---------------------------------------------------------------------------
extern "C" void kernel_launch(void* const* d_in, const int* in_sizes, int n_in,
                              void* d_out, int out_size, void* d_ws, size_t ws_size,
                              hipStream_t stream) {
  const float* q  = (const float*)d_in[0];
  const float* k  = (const float*)d_in[1];
  const float* v  = (const float*)d_in[2];
  const float* Wq = (const float*)d_in[3];
  const float* Wk = (const float*)d_in[4];
  const float* Wv = (const float*)d_in[5];
  const float* bq = (const float*)d_in[6];
  const float* bk = (const float*)d_in[7];
  const float* bv = (const float*)d_in[8];

  char* ws = (char*)d_ws;
  bf16_t* xbf = (bf16_t*)(ws + XBF_OFF);
  bf16_t* wt  = (bf16_t*)(ws + WT_OFF);
  bf16_t* Qh  = (bf16_t*)(ws + Q_OFF);
  bf16_t* Kh  = (bf16_t*)(ws + K_OFF);
  bf16_t* Vt  = (bf16_t*)(ws + V_OFF);
  float* out = (float*)d_out;

  cvt_inputs_kernel<<<6144, 256, 0, stream>>>(q, k, v, xbf);       // 12.58M elems
  cvt_weights_kernel<<<3072, 256, 0, stream>>>(Wq, Wk, Wv, wt);    // 786K elems
  proj_kernel<<<1536, 128, 0, stream>>>(xbf, wt, bq, bk, bv, Qh, Kh, Vt); // 6144 waves
  attn_kernel<<<1024, 128, 0, stream>>>(Qh, Kh, Vt, out);          // 4096 waves
}
